// TTS_Simple_81578608820358
// MI455X (gfx1250) — matrix operand order repaired and checked by reading, not compiled, not run on hardware
//
#include <hip/hip_runtime.h>
#include <math.h>

// ---------------------------------------------------------------------------
// Problem constants
// ---------------------------------------------------------------------------
#define Bn    64
#define TT    256
#define TM    600
#define Ee    512
#define Hh    1024
#define Mb    80
#define KENC  1536            // E + H
#define KDEC  1120            // pad(Mb->96) + H (multiple of 32)
#define NPRJ  96              // proj(80) + gate(1) + pad(15)
#define NBLK  64              // persistent grid size
#define KSTR  136             // LDS A row stride in ushorts (bank-conflict pad)
#define KT_ENC 48
#define KT_DEC 35
#define KT_PRJ 32
#define TILE_US 512           // ushorts per swizzled B tile (32 lanes x 16)
#define NC_ENC 12             // 1536/128 k-chunks
#define NC_DEC 9              // 8*128 + 96
#define NC_PRJ 8              // 1024/128
#define LABUF  (64 * KSTR)    // one LDS A buffer (ushorts)

// ---------------------------------------------------------------------------
// Types
// ---------------------------------------------------------------------------
typedef __attribute__((ext_vector_type(16))) __bf16          v16bf;
typedef __attribute__((ext_vector_type(8)))  float           v8f;
typedef __attribute__((ext_vector_type(16))) unsigned short  v16u;
typedef __attribute__((ext_vector_type(8)))  unsigned short  v8u;

__device__ __forceinline__ v8f wmma_bf16(v16bf a, v16bf b, v8f c) {
  return __builtin_amdgcn_wmma_f32_16x16x32_bf16(
      false, a, false, b, (short)0, c, false, false);
}

__device__ __forceinline__ unsigned short f2bf(float f) {
  unsigned u = __builtin_bit_cast(unsigned, f);
  unsigned r = u + 0x7fffu + ((u >> 16) & 1u);
  return (unsigned short)(r >> 16);
}

__device__ __forceinline__ float sigm(float x) {
  return 1.0f / (1.0f + __expf(-x));
}

__device__ __forceinline__ v16bf load_b_tile(const unsigned short* tileBase,
                                             int lane) {
  return __builtin_bit_cast(v16bf, *(const v16u*)(tileBase + lane * 16));
}

__device__ __forceinline__ v16bf load_a_tile(const unsigned short* lA,
                                             int rowBase, int lane, int k0) {
  const int m   = lane & 15;
  const int sel = lane >> 4;
  const unsigned short* ap = lA + (rowBase + m) * KSTR + k0 + sel * 8;
  v8u lo = *(const v8u*)(ap);
  v8u hi = *(const v8u*)(ap + 16);
  v16u r = __builtin_shufflevector(lo, hi, 0,1,2,3,4,5,6,7,8,9,10,11,12,13,14,15);
  return __builtin_bit_cast(v16bf, r);
}

// 16B global->LDS async DMA granule (ASYNCcnt-tracked)
__device__ __forceinline__ void stage16(unsigned short* ldst,
                                        const unsigned short* gsrc) {
  unsigned lds_addr = (unsigned)(size_t)(void*)ldst;
  asm volatile("global_load_async_to_lds_b128 %0, %1, off"
               :: "v"(lds_addr), "v"(gsrc) : "memory");
}
__device__ __forceinline__ void wait_async_le4() {
  asm volatile("s_wait_asynccnt 0x4" ::: "memory");
}
__device__ __forceinline__ void wait_async_le0() {
  asm volatile("s_wait_asynccnt 0x0" ::: "memory");
}

// ---------------------------------------------------------------------------
// Chunk-copy issue helpers (4 async b128 per wave per chunk)
// ---------------------------------------------------------------------------
__device__ __forceinline__ void enc_issue(unsigned short* buf,
                                          const unsigned short* exrow,
                                          const unsigned short* hbin,
                                          int kc0, int tid) {
#pragma unroll
  for (int u = tid; u < 1024; u += 256) {
    int b = u >> 4, s = u & 15;
    const unsigned short* gs =
        (kc0 < Ee) ? (exrow + (size_t)b * Ee + kc0 + s * 8)
                   : (hbin + (size_t)b * Hh + (kc0 - Ee) + s * 8);
    stage16(&buf[b * KSTR + s * 8], gs);
  }
}

__device__ __forceinline__ void dec_issue(unsigned short* buf,
                                          const unsigned short* melbf,
                                          const unsigned short* hbin,
                                          int kc0, int kch, int tid) {
#pragma unroll
  for (int u = tid; u < 1024; u += 256) {
    int b = u >> 4, s = u & 15;
    if (s * 8 < kch) {
      int gk0 = kc0 + s * 8;
      const unsigned short* gs =
          (gk0 < NPRJ) ? (melbf + b * NPRJ + gk0)
                       : (hbin + (size_t)b * Hh + (gk0 - NPRJ));
      stage16(&buf[b * KSTR + s * 8], gs);
    }
  }
}

__device__ __forceinline__ void h_issue(unsigned short* buf,
                                        const unsigned short* hsrc,
                                        int kc0, int tid) {
#pragma unroll
  for (int u = tid; u < 1024; u += 256) {
    int b = u >> 4, s = u & 15;
    stage16(&buf[b * KSTR + s * 8], hsrc + (size_t)b * Hh + kc0 + s * 8);
  }
}

// ---------------------------------------------------------------------------
// Device-scope generation barrier; waiters use CDNA5 MWAIT (monitor + sleep)
// ---------------------------------------------------------------------------
__device__ __forceinline__ void grid_barrier(unsigned* cnt, unsigned* gen) {
  __threadfence();
  __syncthreads();
  if (threadIdx.x == 0) {
    unsigned g = __hip_atomic_load(gen, __ATOMIC_ACQUIRE, __HIP_MEMORY_SCOPE_AGENT);
    unsigned a = __hip_atomic_fetch_add(cnt, 1u, __ATOMIC_ACQ_REL, __HIP_MEMORY_SCOPE_AGENT);
    if (a == (unsigned)(NBLK - 1)) {
      __hip_atomic_store(cnt, 0u, __ATOMIC_RELAXED, __HIP_MEMORY_SCOPE_AGENT);
      __hip_atomic_fetch_add(gen, 1u, __ATOMIC_RELEASE, __HIP_MEMORY_SCOPE_AGENT);
    } else {
      while (__hip_atomic_load(gen, __ATOMIC_ACQUIRE, __HIP_MEMORY_SCOPE_AGENT) == g) {
        unsigned cur;
        // Arm L2 cacheline monitor on gen, then sleep until it is written.
        asm volatile("global_load_monitor_b32 %0, %1, off scope:SCOPE_DEV\n\t"
                     "s_wait_loadcnt 0x0"
                     : "=v"(cur) : "v"(gen) : "memory");
        if (cur != g) break;
        asm volatile("s_monitor_sleep 0x1" ::: "memory");
      }
    }
  }
  __syncthreads();
}

// ---------------------------------------------------------------------------
// Prep: gather embeddings to bf16, swizzle weights into WMMA tile order.
// Tile order (gates):  [blk][kt][gate][lane][16]  (1KB tiles, lane*32B)
// Tile order (proj):   [kt][nt][lane][16]
// ---------------------------------------------------------------------------
__global__ void prep_kernel(const int* __restrict__ x,
                            const float* __restrict__ emb,
                            const float* __restrict__ encWih,
                            const float* __restrict__ encWhh,
                            const float* __restrict__ decWih,
                            const float* __restrict__ decWhh,
                            const float* __restrict__ projW,
                            const float* __restrict__ gateW,
                            unsigned short* __restrict__ wtE,
                            unsigned short* __restrict__ wtD,
                            unsigned short* __restrict__ wtP,
                            unsigned short* __restrict__ exbf,
                            unsigned short* __restrict__ hbf,
                            float* __restrict__ cbuf,
                            unsigned short* __restrict__ melbf,
                            unsigned* __restrict__ bar) {
  const long i0 = (long)blockIdx.x * blockDim.x + threadIdx.x;
  const long st = (long)gridDim.x * blockDim.x;

  for (long i = i0; i < (long)TT * Bn * Ee; i += st) {
    int e = (int)(i & (Ee - 1));
    int b = (int)((i >> 9) & 63);
    int t = (int)(i >> 15);
    exbf[i] = f2bf(emb[(size_t)x[b * TT + t] * Ee + e]);
  }
  for (long i = i0; i < 64L * KT_ENC * 4 * TILE_US; i += st) {
    int e    = (int)(i & 15);
    int lane = (int)((i >> 4) & 31);
    int ga   = (int)((i >> 9) & 3);
    long rem = i >> 11;
    int kt   = (int)(rem % KT_ENC);
    int blk  = (int)(rem / KT_ENC);
    int k = kt * 32 + (e & 7) + ((lane >> 4) << 3) + ((e >> 3) << 4);
    int n = ga * Hh + blk * 16 + (lane & 15);
    float v = (k < Ee) ? encWih[(size_t)n * Ee + k]
                       : encWhh[(size_t)n * Hh + (k - Ee)];
    wtE[i] = f2bf(v);
  }
  for (long i = i0; i < 64L * KT_DEC * 4 * TILE_US; i += st) {
    int e    = (int)(i & 15);
    int lane = (int)((i >> 4) & 31);
    int ga   = (int)((i >> 9) & 3);
    long rem = i >> 11;
    int kt   = (int)(rem % KT_DEC);
    int blk  = (int)(rem / KT_DEC);
    int k = kt * 32 + (e & 7) + ((lane >> 4) << 3) + ((e >> 3) << 4);
    int n = ga * Hh + blk * 16 + (lane & 15);
    float v;
    if (k < NPRJ) v = (k < Mb) ? decWih[(size_t)n * Mb + k] : 0.0f;
    else          v = decWhh[(size_t)n * Hh + (k - NPRJ)];
    wtD[i] = f2bf(v);
  }
  for (long i = i0; i < (long)KT_PRJ * 6 * TILE_US; i += st) {
    int e    = (int)(i & 15);
    int lane = (int)((i >> 4) & 31);
    long rem = i >> 9;
    int nt   = (int)(rem % 6);
    int kt   = (int)(rem / 6);
    int k = kt * 32 + (e & 7) + ((lane >> 4) << 3) + ((e >> 3) << 4);
    int n = nt * 16 + (lane & 15);
    float v = (n < Mb) ? projW[(size_t)n * Hh + k]
                       : ((n == Mb) ? gateW[k] : 0.0f);
    wtP[i] = f2bf(v);
  }
  for (long i = i0; i < 2L * Bn * Hh; i += st) hbf[i] = 0;
  for (long i = i0; i < (long)Bn * Hh;  i += st) cbuf[i] = 0.0f;
  for (long i = i0; i < (long)Bn * NPRJ; i += st) melbf[i] = 0;
  if (i0 == 0) { bar[0] = 0u; bar[1] = 0u; }
}

// ---------------------------------------------------------------------------
// Encoder: persistent, 64 blocks; double-buffered async LDS staging.
// ---------------------------------------------------------------------------
__global__ __launch_bounds__(256) void enc_kernel(
    const int* __restrict__ tlens, const float* __restrict__ enc_b,
    const unsigned short* __restrict__ wtE,
    const unsigned short* __restrict__ exbf,
    unsigned short* __restrict__ hbf,
    float* __restrict__ cbuf, unsigned* __restrict__ bar) {
  __shared__ __align__(32) unsigned short lA[2 * LABUF];
  __shared__ float lgs[64 * 68];
  __shared__ int   stl[64];
  __shared__ float bia[64];

  const int tid  = threadIdx.x;
  const int lane = tid & 31;
  const int w    = tid >> 5;
  const int j0   = blockIdx.x * 16;
  const int ga   = w & 3;
  const int mgrp = w >> 2;

  if (tid < 64) {
    stl[tid] = tlens[tid];
    bia[tid] = enc_b[(tid >> 4) * Hh + j0 + (tid & 15)];
  }
  unsigned* bcnt = bar;
  unsigned* bgen = bar + 1;
  const size_t blkTile = (size_t)blockIdx.x * KT_ENC;

  for (int t = 0; t < TT; ++t) {
    const int p = t & 1;
    const unsigned short* hbin  = hbf + (size_t)p * Bn * Hh;
    unsigned short*       hbout = hbf + (size_t)(1 - p) * Bn * Hh;
    const unsigned short* exrow = exbf + (size_t)t * Bn * Ee;

    v8f acc0 = {}; v8f acc1 = {};

    enc_issue(lA, exrow, hbin, 0, tid);              // chunk 0 -> buf 0
    for (int c = 0; c < NC_ENC; ++c) {
      const unsigned short* cur = lA + (c & 1) * LABUF;
      __syncthreads();                               // next buffer free
      if (c + 1 < NC_ENC) {
        enc_issue(lA + ((c + 1) & 1) * LABUF, exrow, hbin, (c + 1) * 128, tid);
        wait_async_le4();                            // chunk c complete
      } else {
        wait_async_le0();
      }
      __syncthreads();                               // all waves staged

      const int ktbase = c * 4;
#pragma unroll
      for (int ks = 0; ks < 4; ++ks) {
        const int kt = ktbase + ks;
        const unsigned short* bp =
            wtE + (((blkTile + kt) << 2) + ga) * TILE_US;
        if (kt + 4 < KT_ENC)
          __builtin_prefetch(bp + 16 * TILE_US + lane * 16, 0, 3);
        v16bf bmat = load_b_tile(bp, lane);
        v16bf a0 = load_a_tile(cur, mgrp * 32 +  0, lane, ks * 32);
        v16bf a1 = load_a_tile(cur, mgrp * 32 + 16, lane, ks * 32);
        acc0 = wmma_bf16(a0, bmat, acc0);
        acc1 = wmma_bf16(a1, bmat, acc1);
      }
    }

    __syncthreads();
    {
      const int col   = ga * 16 + (lane & 15);
      const int rbase = (lane < 16) ? 0 : 8;
      const int m0    = mgrp * 32;
#pragma unroll
      for (int r = 0; r < 8; ++r) {
        lgs[(m0 + rbase + r) * 68 + col]      = acc0[r];
        lgs[(m0 + 16 + rbase + r) * 68 + col] = acc1[r];
      }
    }
    __syncthreads();

    for (int i = tid; i < 1024; i += 256) {
      int b = i >> 4, jj = i & 15;
      float gi = lgs[b * 68 +      jj] + bia[     jj];
      float gf = lgs[b * 68 + 16 + jj] + bia[16 + jj];
      float gg = lgs[b * 68 + 32 + jj] + bia[32 + jj];
      float go = lgs[b * 68 + 48 + jj] + bia[48 + jj];
      size_t hidx = (size_t)b * Hh + j0 + jj;
      float cold = cbuf[hidx];
      unsigned short holdu = hbin[hidx];
      bool  m  = t < stl[b];
      float cn = m ? (sigm(gf) * cold + sigm(gi) * tanhf(gg)) : cold;
      cbuf[hidx]  = cn;
      hbout[hidx] = m ? f2bf(sigm(go) * tanhf(cn)) : holdu;
    }
    grid_barrier(bcnt, bgen);
  }
}

// ---------------------------------------------------------------------------
// Decoder: persistent autoregressive; fused proj+gate GEMM on blocks 0..2.
// ---------------------------------------------------------------------------
__global__ __launch_bounds__(256) void dec_kernel(
    const int* __restrict__ mlens,
    const float* __restrict__ dec_b, const float* __restrict__ proj_b,
    const float* __restrict__ gate_b,
    const unsigned short* __restrict__ wtD,
    const unsigned short* __restrict__ wtP,
    unsigned short* __restrict__ hbf, float* __restrict__ cbuf,
    unsigned short* __restrict__ melbf,
    float* __restrict__ outMel, float* __restrict__ outGate,
    unsigned* __restrict__ bar) {
  __shared__ __align__(32) unsigned short lA[2 * LABUF];
  __shared__ float lgs[64 * 68];
  __shared__ int   sml[64];
  __shared__ float bia[64];

  const int tid  = threadIdx.x;
  const int lane = tid & 31;
  const int w    = tid >> 5;
  const int j0   = blockIdx.x * 16;
  const int ga   = w & 3;
  const int mgrp = w >> 2;

  if (tid < 64) {
    sml[tid] = mlens[tid];
    bia[tid] = dec_b[(tid >> 4) * Hh + j0 + (tid & 15)];
  }
  unsigned* bcnt = bar;
  unsigned* bgen = bar + 1;
  const size_t blkTile = (size_t)blockIdx.x * KT_DEC;

  for (int t = 0; t < TM; ++t) {
    const int p = t & 1;
    const unsigned short* hbin  = hbf + (size_t)p * Bn * Hh;
    unsigned short*       hbout = hbf + (size_t)(1 - p) * Bn * Hh;

    v8f acc0 = {}; v8f acc1 = {};

    dec_issue(lA, melbf, hbin, 0, 128, tid);          // chunk 0 -> buf 0
    for (int c = 0; c < NC_DEC; ++c) {
      const unsigned short* cur = lA + (c & 1) * LABUF;
      const int kch = (c == NC_DEC - 1) ? 96 : 128;
      __syncthreads();
      if (c + 1 < NC_DEC) {
        const int kchn = (c + 1 == NC_DEC - 1) ? 96 : 128;
        dec_issue(lA + ((c + 1) & 1) * LABUF, melbf, hbin,
                  (c + 1) * 128, kchn, tid);
        wait_async_le4();
      } else {
        wait_async_le0();
      }
      __syncthreads();

      const int ktbase = c * 4;
      const int ksteps = kch >> 5;
      for (int ks = 0; ks < ksteps; ++ks) {
        const int kt = ktbase + ks;
        const unsigned short* bp =
            wtD + (((blkTile + kt) << 2) + ga) * TILE_US;
        if (kt + 4 < KT_DEC)
          __builtin_prefetch(bp + 16 * TILE_US + lane * 16, 0, 3);
        v16bf bmat = load_b_tile(bp, lane);
        v16bf a0 = load_a_tile(cur, mgrp * 32 +  0, lane, ks * 32);
        v16bf a1 = load_a_tile(cur, mgrp * 32 + 16, lane, ks * 32);
        acc0 = wmma_bf16(a0, bmat, acc0);
        acc1 = wmma_bf16(a1, bmat, acc1);
      }
    }

    __syncthreads();
    {
      const int col   = ga * 16 + (lane & 15);
      const int rbase = (lane < 16) ? 0 : 8;
      const int m0    = mgrp * 32;
#pragma unroll
      for (int r = 0; r < 8; ++r) {
        lgs[(m0 + rbase + r) * 68 + col]      = acc0[r];
        lgs[(m0 + 16 + rbase + r) * 68 + col] = acc1[r];
      }
    }
    __syncthreads();

    for (int i = tid; i < 1024; i += 256) {
      int b = i >> 4, jj = i & 15;
      float gi = lgs[b * 68 +      jj] + bia[     jj];
      float gf = lgs[b * 68 + 16 + jj] + bia[16 + jj];
      float gg = lgs[b * 68 + 32 + jj] + bia[32 + jj];
      float go = lgs[b * 68 + 48 + jj] + bia[48 + jj];
      size_t hidx = (size_t)b * Hh + j0 + jj;
      float cold = cbuf[hidx];
      float cn = sigm(gf) * cold + sigm(gi) * tanhf(gg);
      cbuf[hidx]  = cn;
      hbout[hidx] = f2bf(sigm(go) * tanhf(cn));
    }
    grid_barrier(bcnt, bgen);   // h(t) visible device-wide

    // ---- fused projection + stop-gate: [64,1024] x [1024,96] ----------
    if (blockIdx.x < 3) {
      v8f pacc = {};
      const int tile = blockIdx.x * 8 + w;   // 0..23
      const int mt = tile / 6, nt = tile % 6;
      h_issue(lA, hbout, 0, tid);
      for (int c = 0; c < NC_PRJ; ++c) {
        const unsigned short* cur = lA + (c & 1) * LABUF;
        __syncthreads();
        if (c + 1 < NC_PRJ) {
          h_issue(lA + ((c + 1) & 1) * LABUF, hbout, (c + 1) * 128, tid);
          wait_async_le4();
        } else {
          wait_async_le0();
        }
        __syncthreads();
        const int ktbase = c * 4;
#pragma unroll
        for (int ks = 0; ks < 4; ++ks) {
          const unsigned short* bp =
              wtP + (size_t)((ktbase + ks) * 6 + nt) * TILE_US;
          v16bf bmat = load_b_tile(bp, lane);
          v16bf amat = load_a_tile(cur, mt * 16, lane, ks * 32);
          pacc = wmma_bf16(amat, bmat, pacc);
        }
      }
      const int rbase = (lane < 16) ? 0 : 8;
      const int n = nt * 16 + (lane & 15);
#pragma unroll
      for (int r = 0; r < 8; ++r) {
        const int b = mt * 16 + rbase + r;
        const float val = pacc[r];
        if (n < Mb) {
          float mv = val + proj_b[n];
          melbf[b * NPRJ + n] = f2bf(mv);             // raw feedback
          outMel[((size_t)b * TM + t) * Mb + n] = (t < sml[b]) ? mv : 0.0f;
        } else if (n == Mb) {
          float gv = sigm(val + gate_b[0]);
          outGate[(size_t)b * TM + t] = (t < sml[b]) ? gv : 1000.0f;
        }
      }
    }
    grid_barrier(bcnt, bgen);   // mel(t) feedback visible before t+1
  }
}

// ---------------------------------------------------------------------------
// Host entry
// ---------------------------------------------------------------------------
extern "C" void kernel_launch(void* const* d_in, const int* in_sizes, int n_in,
                              void* d_out, int out_size, void* d_ws,
                              size_t ws_size, hipStream_t stream) {
  (void)in_sizes; (void)n_in; (void)out_size; (void)ws_size;

  const int*   x      = (const int*)  d_in[0];
  const int*   tlens  = (const int*)  d_in[1];
  /* d_in[2] = y : unused */
  const int*   mlens  = (const int*)  d_in[3];
  const float* emb    = (const float*)d_in[4];
  const float* encWih = (const float*)d_in[5];
  const float* encWhh = (const float*)d_in[6];
  const float* encB   = (const float*)d_in[7];
  const float* decWih = (const float*)d_in[8];
  const float* decWhh = (const float*)d_in[9];
  const float* decB   = (const float*)d_in[10];
  const float* projW  = (const float*)d_in[11];
  const float* projB  = (const float*)d_in[12];
  const float* gateW  = (const float*)d_in[13];
  const float* gateB  = (const float*)d_in[14];

  char* ws = (char*)d_ws;
  constexpr size_t SZ_WTE = 64ULL * KT_ENC * 4 * TILE_US * 2;
  constexpr size_t SZ_WTD = 64ULL * KT_DEC * 4 * TILE_US * 2;
  constexpr size_t SZ_WTP = (size_t)KT_PRJ * 6 * TILE_US * 2;
  constexpr size_t SZ_EX  = (size_t)TT * Bn * Ee * 2;
  constexpr size_t OFF_BAR = 0;
  constexpr size_t OFF_WTE = 256;
  constexpr size_t OFF_WTD = OFF_WTE + SZ_WTE;
  constexpr size_t OFF_WTP = OFF_WTD + SZ_WTD;
  constexpr size_t OFF_EX  = OFF_WTP + SZ_WTP;
  constexpr size_t OFF_HBF = OFF_EX  + SZ_EX;
  constexpr size_t OFF_C   = OFF_HBF + 2ULL * Bn * Hh * 2;
  constexpr size_t OFF_MEL = OFF_C   + (size_t)Bn * Hh * 4;

  unsigned*       bar  = (unsigned*)      (ws + OFF_BAR);
  unsigned short* wtE  = (unsigned short*)(ws + OFF_WTE);
  unsigned short* wtD  = (unsigned short*)(ws + OFF_WTD);
  unsigned short* wtP  = (unsigned short*)(ws + OFF_WTP);
  unsigned short* exbf = (unsigned short*)(ws + OFF_EX);
  unsigned short* hbf  = (unsigned short*)(ws + OFF_HBF);
  float*          cbuf = (float*)         (ws + OFF_C);
  unsigned short* melb = (unsigned short*)(ws + OFF_MEL);

  float* outMel  = (float*)d_out;
  float* outGate = outMel + (size_t)Bn * TM * Mb;

  prep_kernel<<<2048, 256, 0, stream>>>(x, emb, encWih, encWhh, decWih, decWhh,
                                        projW, gateW, wtE, wtD, wtP,
                                        exbf, hbf, cbuf, melb, bar);
  enc_kernel<<<NBLK, 256, 0, stream>>>(tlens, encB, wtE, exbf, hbf, cbuf, bar);
  dec_kernel<<<NBLK, 256, 0, stream>>>(mlens, decB, projB, gateB, wtD, wtP,
                                       hbf, cbuf, melb, outMel, outGate, bar);
}
